// Position_attention_67525475827870
// MI455X (gfx1250) — matrix operand order repaired and checked by reading, not compiled, not run on hardware
//
#include <hip/hip_runtime.h>
#include <math.h>

typedef __attribute__((ext_vector_type(16))) _Float16 v16h;
typedef __attribute__((ext_vector_type(4)))  _Float16 v4h;
typedef __attribute__((ext_vector_type(8)))  float    v8f;
typedef __attribute__((ext_vector_type(4)))  float    v4f;

#define N_PIX 4096
#define C_DIM 512
#define D_DIM 64

static __device__ __forceinline__ v8f wmma32(v16h a, v16h b, v8f c) {
  // v_wmma_f32_16x16x32_f16: D = A(16x32 f16) * B(32x16 f16) + C(16x16 f32)
  return __builtin_amdgcn_wmma_f32_16x16x32_f16(false, a, false, b, (short)0, c, false, false);
}

// A-fragment (16-bit A 16x32): per lane, halves 0..7 -> K = 16*half + t,
// halves 8..15 -> K = 16*half + 8 + t. Caller passes the two 8-element chunks.
static __device__ __forceinline__ v16h load_a_f16chunks(const _Float16* p0, const _Float16* p1) {
  v16h r;
#pragma unroll
  for (int t = 0; t < 8; ++t) { r[t] = p0[t]; r[t + 8] = p1[t]; }
  return r;
}
// B-fragment (16-bit B 32x16): lanes 0-15 hold K=0..15, lanes 16-31 hold K=16..31,
// contiguous per lane -> one 32-byte run at (base + 16*half), base = column n of B
// stored column-major (i.e. row n of B^T, row-major).
static __device__ __forceinline__ v16h load_b_f16(const _Float16* p) {
  return *(const v16h*)p;
}

// ---------------- x f32 -> f16, vectorized 4-wide ----------------
__global__ __launch_bounds__(256) void cvt_x_f16(const float* __restrict__ X,
                                                 _Float16* __restrict__ Xh, int n4) {
  int i = blockIdx.x * blockDim.x + threadIdx.x;
  if (i >= n4) return;
  v4f v = ((const v4f*)X)[i];
  v4h h;
#pragma unroll
  for (int t = 0; t < 4; ++t) h[t] = (_Float16)v[t];
  ((v4h*)Xh)[i] = h;
}

// ---------------- weight transpose+convert: Wt[n][k] = (f16)W[k][n] ----------------
__global__ void transpose_w(const float* __restrict__ W, _Float16* __restrict__ Wt,
                            int K, int Nc) {
  int idx = blockIdx.x * blockDim.x + threadIdx.x;
  if (idx >= K * Nc) return;
  int n = idx / K;
  int k = idx - n * K;
  Wt[idx] = (_Float16)W[(size_t)k * Nc + n];
}

// ---------------- projection GEMM: Y(16x64 strip) = X[16x512] * Wt^T ----------------
// one wave per 16-row x 64-col strip: A-fragment loaded once per K step, 4 WMMAs.
// Xh is f16 [16384, 512]; Wt is f16 [ncols, 512].
// transposed==0: Y row-major [rows, ncols] f16 (q, k)
// transposed==1: Y = vt[b][c][npix] f16 (v, stored transposed for attention)
__global__ __launch_bounds__(32) void proj_gemm(const _Float16* __restrict__ Xh,
                                                const _Float16* __restrict__ Wt,
                                                _Float16* __restrict__ Y,
                                                int ncols, int transposed) {
  const int lane = threadIdx.x & 31;
  const int half = lane >> 4;
  const int mn   = lane & 15;
  const int row0 = blockIdx.x * 16;
  const int col0 = blockIdx.y * 64;

  const _Float16* xrow = Xh + (size_t)(row0 + mn) * C_DIM;
  const _Float16* wrow = Wt + (size_t)(col0 + mn) * C_DIM;

  v8f acc[4];
#pragma unroll
  for (int c = 0; c < 4; ++c) acc[c] = (v8f){0.f, 0.f, 0.f, 0.f, 0.f, 0.f, 0.f, 0.f};

#pragma unroll 4
  for (int kk = 0; kk < C_DIM; kk += 32) {
    v16h a = load_a_f16chunks(xrow + kk + 16 * half, xrow + kk + 8 + 16 * half);
#pragma unroll
    for (int c = 0; c < 4; ++c) {
      acc[c] = wmma32(a, load_b_f16(wrow + (size_t)c * 16 * C_DIM + kk + 16 * half), acc[c]);
    }
  }

  if (!transposed) {
#pragma unroll
    for (int c = 0; c < 4; ++c) {
#pragma unroll
      for (int r = 0; r < 8; ++r) {
        int m = r + 8 * half;  // D layout: vgpr r holds row m = r + 8*(lane>=16)
        Y[(size_t)(row0 + m) * ncols + col0 + c * 16 + mn] = (_Float16)acc[c][r];
      }
    }
  } else {
    int b  = row0 >> 12;              // 4096 pixels per batch
    int np = row0 & (N_PIX - 1);
    _Float16* yb = Y + (size_t)b * C_DIM * N_PIX + np;
#pragma unroll
    for (int c = 0; c < 4; ++c) {
#pragma unroll
      for (int r = 0; r < 8; ++r) {
        int m = r + 8 * half;
        yb[(size_t)(col0 + c * 16 + mn) * N_PIX + m] = (_Float16)acc[c][r];
      }
    }
  }
}

// ---------------- flash attention ----------------
// energy[i,j] = k_i . q_j ; softmax over j ; out_i = sum_j p_ij v_j ; out *= gamma
// 1 block (4 waves) per (batch, 16-row i tile). Wave w owns V columns [128w,128w+128).
__global__ __launch_bounds__(128) void flash_attn(const _Float16* __restrict__ qf,
                                                  const _Float16* __restrict__ kf,
                                                  const _Float16* __restrict__ vt,
                                                  const float* __restrict__ gamma,
                                                  float* __restrict__ out) {
  __shared__ _Float16 ptile[4][16][40];  // per-wave 16x32 P tile, padded rows (80B)

  const int wave = threadIdx.x >> 5;
  const int lane = threadIdx.x & 31;
  const int half = lane >> 4;
  const int mn   = lane & 15;
  const int b    = blockIdx.x >> 8;          // 256 i-tiles per batch
  const int i0   = (blockIdx.x & 255) * 16;
  const int c0   = wave * 128;

  // K rows (attention "queries"): A-fragments over depth d=64 (two K=32 chunks)
  const _Float16* krow = kf + (size_t)(b * N_PIX + i0 + mn) * D_DIM;
  const v16h a_k0 = load_a_f16chunks(krow +      16 * half, krow +  8 + 16 * half);
  const v16h a_k1 = load_a_f16chunks(krow + 32 + 16 * half, krow + 40 + 16 * half);

  v8f acc[8];
#pragma unroll
  for (int t = 0; t < 8; ++t) acc[t] = (v8f){0.f, 0.f, 0.f, 0.f, 0.f, 0.f, 0.f, 0.f};

  float mrow[8], lrow[8];
#pragma unroll
  for (int r = 0; r < 8; ++r) { mrow[r] = -INFINITY; lrow[r] = 0.f; }

  const _Float16* qbase = qf + (size_t)b * N_PIX * D_DIM;
  const _Float16* vbase = vt + (size_t)b * C_DIM * N_PIX;

  for (int j0 = 0; j0 < N_PIX; j0 += 32) {
    // S = K_i * Q_j^T, 16 rows x 32 j-cols (two 16x16 D fragments)
    const _Float16* q0 = qbase + (size_t)(j0 + mn) * D_DIM;
    const _Float16* q1 = qbase + (size_t)(j0 + 16 + mn) * D_DIM;
    if (j0 + 32 < N_PIX)  // gfx1250 global_prefetch_b8 of next q window
      __builtin_prefetch(qbase + (size_t)(j0 + 32 + mn) * D_DIM, 0, 1);
    v8f s0 = (v8f){0.f, 0.f, 0.f, 0.f, 0.f, 0.f, 0.f, 0.f};
    v8f s1 = (v8f){0.f, 0.f, 0.f, 0.f, 0.f, 0.f, 0.f, 0.f};
    s0 = wmma32(a_k0, load_b_f16(q0 +      16 * half), s0);
    s0 = wmma32(a_k1, load_b_f16(q0 + 32 + 16 * half), s0);
    s1 = wmma32(a_k0, load_b_f16(q1 +      16 * half), s1);
    s1 = wmma32(a_k1, load_b_f16(q1 + 32 + 16 * half), s1);

    // online softmax: D layout puts row m = r + 8*half on vgpr r, cols across the
    // 16-lane half -> xor shuffles 1/2/4/8 reduce exactly one softmax row.
    v8f scale;
#pragma unroll
    for (int r = 0; r < 8; ++r) {
      float mx = fmaxf(s0[r], s1[r]);
      mx = fmaxf(mx, __shfl_xor(mx, 1));
      mx = fmaxf(mx, __shfl_xor(mx, 2));
      mx = fmaxf(mx, __shfl_xor(mx, 4));
      mx = fmaxf(mx, __shfl_xor(mx, 8));
      float mnew = fmaxf(mrow[r], mx);
      float sc = __expf(mrow[r] - mnew);
      float p0 = __expf(s0[r] - mnew);
      float p1 = __expf(s1[r] - mnew);
      float sum = p0 + p1;
      sum += __shfl_xor(sum, 1);
      sum += __shfl_xor(sum, 2);
      sum += __shfl_xor(sum, 4);
      sum += __shfl_xor(sum, 8);
      lrow[r] = lrow[r] * sc + sum;
      mrow[r] = mnew;
      scale[r] = sc;
      int m = r + 8 * half;
      ptile[wave][m][mn]      = (_Float16)p0;   // D layout -> LDS (row-major P)
      ptile[wave][m][mn + 16] = (_Float16)p1;
    }
#pragma unroll
    for (int t = 0; t < 8; ++t) acc[t] *= scale;

    // reload P as an A-fragment (per-wave tile; LDS is in-order within a wave)
    v16h a_p = load_a_f16chunks(&ptile[wave][mn][16 * half],
                                &ptile[wave][mn][8 + 16 * half]);

    // acc += P(16x32) * V(32x128): 8 WMMAs over this wave's V columns
#pragma unroll
    for (int t = 0; t < 8; ++t) {
      const _Float16* vrow = vbase + (size_t)(c0 + t * 16 + mn) * N_PIX + j0;
      acc[t] = wmma32(a_p, load_b_f16(vrow + 16 * half), acc[t]);
    }
  }

  const float g = gamma[0];
  float* obase = out + (size_t)b * N_PIX * C_DIM + (size_t)i0 * C_DIM;
#pragma unroll
  for (int t = 0; t < 8; ++t) {
#pragma unroll
    for (int r = 0; r < 8; ++r) {
      int m = r + 8 * half;
      obase[(size_t)m * C_DIM + c0 + t * 16 + mn] = g * acc[t][r] / lrow[r];
    }
  }
}

extern "C" void kernel_launch(void* const* d_in, const int* in_sizes, int n_in,
                              void* d_out, int out_size, void* d_ws, size_t ws_size,
                              hipStream_t stream) {
  (void)in_sizes; (void)n_in; (void)out_size; (void)ws_size;
  const float* x     = (const float*)d_in[0];   // [4,64,64,512] f32
  const float* Wq    = (const float*)d_in[1];   // [512,64]
  const float* Wk    = (const float*)d_in[2];   // [512,64]
  const float* Wv    = (const float*)d_in[3];   // [512,512]
  const float* gamma = (const float*)d_in[4];   // [1]
  float* out = (float*)d_out;                   // [4,64,64,512] f32

  // workspace layout (f16 elements)
  _Float16* ws  = (_Float16*)d_ws;
  const size_t NQ = (size_t)4 * N_PIX * D_DIM;           // 1,048,576
  _Float16* qf  = ws;                                    // [4*4096, 64]
  _Float16* kf  = qf + NQ;                               // [4*4096, 64]
  _Float16* vt  = kf + NQ;                               // [4, 512, 4096]
  _Float16* wtq = vt + (size_t)4 * C_DIM * N_PIX;        // [64, 512]
  _Float16* wtk = wtq + (size_t)D_DIM * C_DIM;           // [64, 512]
  _Float16* wtv = wtk + (size_t)D_DIM * C_DIM;           // [512, 512]

  // xh (16 MB) lives in d_out (32 MB): d_out is dead until flash_attn's final
  // stores, which happen strictly after the projections finish reading xh.
  _Float16* xh = (_Float16*)d_out;                       // [16384, 512] f16

  // 1) x -> f16 once (removes all redundant f32 loads + converts in the GEMMs)
  const int n4 = (16384 * C_DIM) / 4;
  cvt_x_f16<<<(n4 + 255) / 256, 256, 0, stream>>>(x, xh, n4);

  // 2) f16 transposed weights
  transpose_w<<<(512 * 64 + 255) / 256, 256, 0, stream>>>(Wq, wtq, 512, 64);
  transpose_w<<<(512 * 64 + 255) / 256, 256, 0, stream>>>(Wk, wtk, 512, 64);
  transpose_w<<<(512 * 512 + 255) / 256, 256, 0, stream>>>(Wv, wtv, 512, 512);

  // 3) projections (one wave per 16x64 strip, 4-way A-fragment reuse)
  dim3 gq(1024, 1), gv(1024, 8);
  proj_gemm<<<gq, 32, 0, stream>>>(xh, wtq, qf, D_DIM, 0);
  proj_gemm<<<gq, 32, 0, stream>>>(xh, wtk, kf, D_DIM, 0);
  proj_gemm<<<gv, 32, 0, stream>>>(xh, wtv, vt, C_DIM, 1);

  // 4) fused flash attention + gamma (overwrites all of d_out)
  flash_attn<<<4 * (N_PIX / 16), 128, 0, stream>>>(qf, kf, vt, gamma, out);
}
